// Intra_GCN_61967788146853
// MI455X (gfx1250) — matrix operand order repaired and checked by reading, not compiled, not run on hardware
//
#include <hip/hip_runtime.h>

#define IN_F  512
#define HID_F 256
#define OUT_F 128
#define LN_EPS 1e-5f

typedef __attribute__((ext_vector_type(16))) __bf16 v16bf;
typedef __attribute__((ext_vector_type(8)))  __bf16 v8bf;
typedef __attribute__((ext_vector_type(8)))  float  v8f;

static __device__ __forceinline__ v16bf cat8(v8bf lo, v8bf hi) {
  v16bf r;
#pragma unroll
  for (int i = 0; i < 8; ++i) { r[i] = lo[i]; r[8 + i] = hi[i]; }
  return r;
}

// ---------------- small utility kernels ----------------

__global__ void zero_scal_kernel(float* scal) {
  if (threadIdx.x < 64) scal[threadIdx.x] = 0.0f;
}

__global__ void fill_one_kernel(float* __restrict__ p, int n) {
  int i = blockIdx.x * blockDim.x + threadIdx.x;
  if (i < n) p[i] = 1.0f;
}

__global__ void deg_acc_kernel(const int* __restrict__ dst, float* __restrict__ deg, int e) {
  int i = blockIdx.x * blockDim.x + threadIdx.x;
  if (i < e) atomicAdd(&deg[dst[i]], 1.0f);
}

__global__ void rsqrt_kernel(float* __restrict__ p, int n) {
  int i = blockIdx.x * blockDim.x + threadIdx.x;
  if (i < n) p[i] = rsqrtf(p[i]);
}

// sum / sumsq reduction (atomic accumulate into scalars; caller zeroes first)
__global__ __launch_bounds__(256) void stats_kernel(const float* __restrict__ x, long n,
                                                    float* __restrict__ sum, float* __restrict__ sq) {
  __shared__ float ss[256], sz[256];
  float s = 0.f, q = 0.f;
  for (long i = blockIdx.x * (long)blockDim.x + threadIdx.x; i < n;
       i += (long)gridDim.x * blockDim.x) {
    float v = x[i]; s += v; q += v * v;
  }
  ss[threadIdx.x] = s; sz[threadIdx.x] = q;
  __syncthreads();
  for (int o = 128; o > 0; o >>= 1) {
    if ((int)threadIdx.x < o) {
      ss[threadIdx.x] += ss[threadIdx.x + o];
      sz[threadIdx.x] += sz[threadIdx.x + o];
    }
    __syncthreads();
  }
  if (threadIdx.x == 0) { atomicAdd(sum, ss[0]); atomicAdd(sq, sz[0]); }
}

// in-place relu fused with stats on the relu'd values
__global__ __launch_bounds__(256) void relu_stats_kernel(float* __restrict__ x, long n,
                                                         float* __restrict__ sum, float* __restrict__ sq) {
  __shared__ float ss[256], sz[256];
  float s = 0.f, q = 0.f;
  for (long i = blockIdx.x * (long)blockDim.x + threadIdx.x; i < n;
       i += (long)gridDim.x * blockDim.x) {
    float v = x[i]; v = v > 0.f ? v : 0.f; x[i] = v;
    s += v; q += v * v;
  }
  ss[threadIdx.x] = s; sz[threadIdx.x] = q;
  __syncthreads();
  for (int o = 128; o > 0; o >>= 1) {
    if ((int)threadIdx.x < o) {
      ss[threadIdx.x] += ss[threadIdx.x + o];
      sz[threadIdx.x] += sz[threadIdx.x + o];
    }
    __syncthreads();
  }
  if (threadIdx.x == 0) { atomicAdd(sum, ss[0]); atomicAdd(sq, sz[0]); }
}

__global__ void finalize_kernel(float* scal, int si, int qi, int mi, int ri, float invcnt) {
  float mean = scal[si] * invcnt;
  float var  = scal[qi] * invcnt - mean * mean;
  scal[mi] = mean;
  scal[ri] = rsqrtf(var + LN_EPS);
}

// (x - mean) * rs * w[c] + b[c]  ->  bf16
__global__ __launch_bounds__(256) void norm_cvt_kernel(const float* __restrict__ in,
                                                       __bf16* __restrict__ out,
                                                       const float* __restrict__ scal, int mi, int ri,
                                                       const float* __restrict__ w,
                                                       const float* __restrict__ b,
                                                       long n, int F) {
  long i = blockIdx.x * (long)blockDim.x + threadIdx.x;
  if (i >= n) return;
  float mean = scal[mi], rs = scal[ri];
  int c = (int)(i % F);
  float v = (in[i] - mean) * rs * w[c] + b[c];
  out[i] = (__bf16)v;
}

// Repack W[K x Nn] (row-major fp32) into bf16 WMMA B-fragment order:
// per 32x16 (KxN) tile, lane-major, 16 bf16 per lane.
__global__ void repack_kernel(const float* __restrict__ W, __bf16* __restrict__ Wp, int K, int Nn) {
  int t = blockIdx.x * blockDim.x + threadIdx.x;
  if (t >= K * Nn) return;
  int k = t / Nn, n = t % Nn;
  int kt = k >> 5, ko = k & 31, nt = n >> 4;
  int lane = ((ko & 8) << 1) + (n & 15);   // lanes 0-15 hold K%32 in {0..7,16..23}, lanes 16-31 in {8..15,24..31}
  int j = (ko & 7) | ((ko & 16) >> 1);     // position within lane's 16 packed bf16 (table order)
  size_t idx = ((size_t)(kt * (Nn >> 4) + nt) * 32 + lane) * 16 + j;
  Wp[idx] = (__bf16)W[t];
}

// ---------------- WMMA GEMM: C[M x Nn] = A[M x K] * B[K x Nn], bf16 in, f32 out ----------------
// Block = 8 waves: 8 m-tiles (128 rows) x one 64-column group.
// The B panel (K x 64, fragment order) is staged once into LDS and shared by all 8 waves;
// the WMMA loop then reads B via ds_load_b128 and streams A with global b128 loads.
__global__ __launch_bounds__(256) void gemm_bf16_wmma_kernel(
    const __bf16* __restrict__ A, const __bf16* __restrict__ Bp,
    float* __restrict__ C, int M, int K, int Nn)
{
  extern __shared__ char smem_raw[];
  __bf16* smem = (__bf16*)smem_raw;

  const int lane   = threadIdx.x & 31;
  const int wavein = threadIdx.x >> 5;          // 0..7
  const int ngroups = Nn >> 6;                  // 64-wide column groups
  const int ntiles  = Nn >> 4;
  const int ktend   = K >> 5;
  const int mtiles  = (M + 15) >> 4;

  const int bm  = blockIdx.x / ngroups;         // row-block (8 m-tiles)
  const int ng  = blockIdx.x % ngroups;
  const int n0  = ng << 6;
  const int nt0 = n0 >> 4;

  // ---- stage B panel into LDS: ktend * 4 tiles of 1KB (fragment order preserved) ----
  {
    uint4* sm4 = (uint4*)smem;
    const uint4* gb = (const uint4*)Bp;
    const int j   = threadIdx.x >> 6;           // tile within group (0..3)
    const int pos = threadIdx.x & 63;           // 16B unit within tile
    for (int kt = 0; kt < ktend; ++kt)
      sm4[(kt * 4 + j) * 64 + pos] = gb[(kt * ntiles + nt0 + j) * 64 + pos];
  }
  __syncthreads();

  const int tm = bm * 8 + wavein;
  if (tm >= mtiles) return;

  // A fragment addressing (16-bit A 16x32 layout):
  // lanes 0-15: row = lane,    K = kt*32 + {0..7, 16..23}
  // lanes16-31: row = lane-16, K = kt*32 + {8..15, 24..31}
  const int row   = (tm << 4) + (lane & 15);
  const int khalf = (lane >> 4) << 3;           // 0 or 8

  v8f acc0 = {}, acc1 = {}, acc2 = {}, acc3 = {};
  for (int kt = 0; kt < ktend; ++kt) {
    const __bf16* ap = A + (size_t)row * K + (kt << 5) + khalf;
    v16bf a = cat8(*(const v8bf*)ap, *(const v8bf*)(ap + 16));

    const __bf16* sb = smem + (((kt * 4) * 32 + lane) << 4);  // +512 elems per n-tile
    v16bf b0 = cat8(*(const v8bf*)(sb +    0), *(const v8bf*)(sb +    8));
    v16bf b1 = cat8(*(const v8bf*)(sb +  512), *(const v8bf*)(sb +  520));
    v16bf b2 = cat8(*(const v8bf*)(sb + 1024), *(const v8bf*)(sb + 1032));
    v16bf b3 = cat8(*(const v8bf*)(sb + 1536), *(const v8bf*)(sb + 1544));

    acc0 = __builtin_amdgcn_wmma_f32_16x16x32_bf16(false, a, false, b0, (short)0, acc0, false, false);
    acc1 = __builtin_amdgcn_wmma_f32_16x16x32_bf16(false, a, false, b1, (short)0, acc1, false, false);
    acc2 = __builtin_amdgcn_wmma_f32_16x16x32_bf16(false, a, false, b2, (short)0, acc2, false, false);
    acc3 = __builtin_amdgcn_wmma_f32_16x16x32_bf16(false, a, false, b3, (short)0, acc3, false, false);
  }

  // D layout: VGPR r -> row = tm*16 + r + 8*(lane>=16), col = n0 + lane%16 (+tile offset)
  const int lh = lane >> 4;
  const int ln = lane & 15;
  float* cb = C + (size_t)((tm << 4) + (lh << 3)) * Nn + n0 + ln;
#pragma unroll
  for (int r = 0; r < 8; ++r) {
    cb[(size_t)r * Nn +  0] = acc0[r];
    cb[(size_t)r * Nn + 16] = acc1[r];
    cb[(size_t)r * Nn + 32] = acc2[r];
    cb[(size_t)r * Nn + 48] = acc3[r];
  }
}

// ---------------- GCN propagation ----------------

// out[i,c] = bias[c] + H[i,c] * dinv[i]^2   (self loop)
__global__ __launch_bounds__(256) void init_out_kernel(const float* __restrict__ H,
                                                       const float* __restrict__ dinv,
                                                       const float* __restrict__ bias,
                                                       float* __restrict__ out, int N, int F) {
  long i = blockIdx.x * (long)blockDim.x + threadIdx.x;
  if (i >= (long)N * F) return;
  int node = (int)(i / F);
  int c = (int)(i % F);
  float di = dinv[node];
  out[i] = bias[c] + H[i] * di * di;
}

// per-edge scatter: out[dst] += H[src] * dinv[src]*dinv[dst], 8 floats per thread
__global__ __launch_bounds__(256) void edge_prop_kernel(const int* __restrict__ src,
                                                        const int* __restrict__ dst,
                                                        const float* __restrict__ dinv,
                                                        const float* __restrict__ H,
                                                        float* __restrict__ out, int E, int F) {
  long t = blockIdx.x * (long)blockDim.x + threadIdx.x;
  int cpe = F >> 3;
  if (t >= (long)E * cpe) return;
  int e = (int)(t / cpe);
  int c = (int)(t % cpe) << 3;
  int s = src[e], d = dst[e];
  float nm = dinv[s] * dinv[d];
  const float4* hp = (const float4*)(H + (size_t)s * F + c);
  float4 h0 = hp[0], h1 = hp[1];
  float* op = out + (size_t)d * F + c;
  atomicAdd(op + 0, h0.x * nm);
  atomicAdd(op + 1, h0.y * nm);
  atomicAdd(op + 2, h0.z * nm);
  atomicAdd(op + 3, h0.w * nm);
  atomicAdd(op + 4, h1.x * nm);
  atomicAdd(op + 5, h1.y * nm);
  atomicAdd(op + 6, h1.z * nm);
  atomicAdd(op + 7, h1.w * nm);
}

// d_out[c] = (colmean(out2[:,c]) - mean) * rs * w[c] + b[c]
__global__ __launch_bounds__(256) void final_out_kernel(const float* __restrict__ out2,
                                                        const float* __restrict__ scal, int mi, int ri,
                                                        const float* __restrict__ w,
                                                        const float* __restrict__ b,
                                                        float* __restrict__ dout, int N, int F) {
  __shared__ float ss[256];
  int c = blockIdx.x;
  float s = 0.f;
  for (int i = threadIdx.x; i < N; i += blockDim.x) s += out2[(size_t)i * F + c];
  ss[threadIdx.x] = s;
  __syncthreads();
  for (int o = 128; o > 0; o >>= 1) {
    if ((int)threadIdx.x < o) ss[threadIdx.x] += ss[threadIdx.x + o];
    __syncthreads();
  }
  if (threadIdx.x == 0)
    dout[c] = (ss[0] / (float)N - scal[mi]) * scal[ri] * w[c] + b[c];
}

// ---------------- host orchestration ----------------

extern "C" void kernel_launch(void* const* d_in, const int* in_sizes, int n_in,
                              void* d_out, int out_size, void* d_ws, size_t ws_size,
                              hipStream_t stream) {
  const float* x    = (const float*)d_in[0];
  const int*   ei   = (const int*)  d_in[1];
  const float* W1   = (const float*)d_in[2];
  const float* b1   = (const float*)d_in[3];
  const float* W2   = (const float*)d_in[4];
  const float* b2   = (const float*)d_in[5];
  const float* ln0w = (const float*)d_in[6];
  const float* ln0b = (const float*)d_in[7];
  const float* ln1w = (const float*)d_in[8];
  const float* ln1b = (const float*)d_in[9];
  const float* ln2w = (const float*)d_in[10];
  const float* ln2b = (const float*)d_in[11];

  const int N = in_sizes[0] / IN_F;
  const int E = in_sizes[1] / 2;
  const int* src = ei;
  const int* dst = ei + E;

  char* ws = (char*)d_ws;
  size_t off = 0;
  auto alloc = [&](size_t bytes) -> size_t {
    size_t o = off;
    off += (bytes + 255) & ~(size_t)255;
    return o;
  };

  float*  scal = (float*) (ws + alloc(64 * sizeof(float)));
  float*  dinv = (float*) (ws + alloc((size_t)N * sizeof(float)));
  __bf16* W1p  = (__bf16*)(ws + alloc((size_t)IN_F * HID_F * 2));
  __bf16* W2p  = (__bf16*)(ws + alloc((size_t)HID_F * OUT_F * 2));
  __bf16* xn   = (__bf16*)(ws + alloc((size_t)N * IN_F * 2));   // reused as h1n
  float*  H    = (float*) (ws + alloc((size_t)N * HID_F * 4));  // reused as H2
  float*  outp = (float*) (ws + alloc((size_t)N * HID_F * 4));  // reused as out2
  __bf16* h1n  = xn;
  float*  H2   = H;
  float*  out2 = outp;

  const long n0e = (long)N * IN_F;
  const long n1e = (long)N * HID_F;
  const long n2e = (long)N * OUT_F;

  // 0. reset scalar accumulators (ws is poisoned between runs)
  zero_scal_kernel<<<1, 64, 0, stream>>>(scal);

  // 1. degrees -> dinv = rsqrt(1 + indeg)
  fill_one_kernel<<<(N + 255) / 256, 256, 0, stream>>>(dinv, N);
  deg_acc_kernel<<<(E + 255) / 256, 256, 0, stream>>>(dst, dinv, E);
  rsqrt_kernel<<<(N + 255) / 256, 256, 0, stream>>>(dinv, N);

  // 2. LN0 stats + normalize+convert x -> bf16
  stats_kernel<<<4096, 256, 0, stream>>>(x, n0e, scal + 0, scal + 1);
  finalize_kernel<<<1, 1, 0, stream>>>(scal, 0, 1, 2, 3, 1.0f / (float)n0e);
  norm_cvt_kernel<<<(int)((n0e + 255) / 256), 256, 0, stream>>>(x, xn, scal, 2, 3, ln0w, ln0b, n0e, IN_F);

  // 3. repack weights into WMMA B-fragment order
  repack_kernel<<<(IN_F * HID_F + 255) / 256, 256, 0, stream>>>(W1, W1p, IN_F, HID_F);
  repack_kernel<<<(HID_F * OUT_F + 255) / 256, 256, 0, stream>>>(W2, W2p, HID_F, OUT_F);

  const int mtiles  = (N + 15) / 16;
  const int mblocks = (mtiles + 7) / 8;

  // 4. GEMM1: H = xn @ W1   (bf16 WMMA, fp32 accumulate; B panel staged in LDS)
  {
    int ngroups = HID_F / 64;
    size_t lds = (size_t)(IN_F / 32) * 4 * 32 * 16 * 2;   // 64 KB
    gemm_bf16_wmma_kernel<<<mblocks * ngroups, 256, lds, stream>>>(xn, W1p, H, N, IN_F, HID_F);
  }

  // 5. propagate layer 1: self-loop + bias, edge atomics, relu + LN1 stats
  init_out_kernel<<<(int)((n1e + 255) / 256), 256, 0, stream>>>(H, dinv, b1, outp, N, HID_F);
  {
    long t = (long)E * (HID_F / 8);
    edge_prop_kernel<<<(int)((t + 255) / 256), 256, 0, stream>>>(src, dst, dinv, H, outp, E, HID_F);
  }
  relu_stats_kernel<<<4096, 256, 0, stream>>>(outp, n1e, scal + 4, scal + 5);
  finalize_kernel<<<1, 1, 0, stream>>>(scal, 4, 5, 6, 7, 1.0f / (float)n1e);
  norm_cvt_kernel<<<(int)((n1e + 255) / 256), 256, 0, stream>>>(outp, h1n, scal, 6, 7, ln1w, ln1b, n1e, HID_F);

  // 6. GEMM2: H2 = h1n @ W2
  {
    int ngroups = OUT_F / 64;
    size_t lds = (size_t)(HID_F / 32) * 4 * 32 * 16 * 2;  // 32 KB
    gemm_bf16_wmma_kernel<<<mblocks * ngroups, 256, lds, stream>>>(h1n, W2p, H2, N, HID_F, OUT_F);
  }

  // 7. propagate layer 2
  init_out_kernel<<<(int)((n2e + 255) / 256), 256, 0, stream>>>(H2, dinv, b2, out2, N, OUT_F);
  {
    long t = (long)E * (OUT_F / 8);
    edge_prop_kernel<<<(int)((t + 255) / 256), 256, 0, stream>>>(src, dst, dinv, H2, out2, E, OUT_F);
  }
  relu_stats_kernel<<<4096, 256, 0, stream>>>(out2, n2e, scal + 8, scal + 9);
  finalize_kernel<<<1, 1, 0, stream>>>(scal, 8, 9, 10, 11, 1.0f / (float)n2e);

  // 8. LN2 + global mean pool -> d_out[128]
  final_out_kernel<<<OUT_F, 256, 0, stream>>>(out2, scal, 10, 11, ln2w, ln2b, (float*)d_out, N, OUT_F);
}
